// LSTM_Model_25323127177386
// MI455X (gfx1250) — matrix operand order repaired and checked by reading, not compiled, not run on hardware
//
#include <hip/hip_runtime.h>
#include <hip/hip_bf16.h>

#define B_TOT   2048
#define T_LEN   512
#define HDIM    64
#define GDIM    256      // 4*H
#define RPB     16       // batch rows per block
#define NTHR    128      // 4 waves (wave32)

typedef _Float16 v16h __attribute__((ext_vector_type(16)));
typedef _Float16 v8h  __attribute__((ext_vector_type(8)));
typedef float    v8f  __attribute__((ext_vector_type(8)));
typedef unsigned int u32x4 __attribute__((ext_vector_type(4)));
typedef int          i32x4 __attribute__((ext_vector_type(4)));
typedef int          i32x8 __attribute__((ext_vector_type(8)));

union V16 {
    v16h v;
    struct Parts { v8h lo; v8h hi; } p;
};

#if defined(__has_builtin)
#if __has_builtin(__builtin_amdgcn_tensor_load_to_lds) && __has_builtin(__builtin_amdgcn_s_wait_tensorcnt)
#define HAVE_TDM 1
#endif
#endif
#ifndef HAVE_TDM
#define HAVE_TDM 0
#endif

#if __has_builtin(__builtin_amdgcn_tanhf)
__device__ __forceinline__ float fast_tanh(float x) {
    return __builtin_amdgcn_tanhf(x);            // v_tanh_f32 (gfx1250 trans op)
}
#else
__device__ __forceinline__ float fast_tanh(float x) {
    float e = __builtin_amdgcn_exp2f(2.885390082f * x);
    return 1.0f - 2.0f * __builtin_amdgcn_rcpf(e + 1.0f);
}
#endif

__device__ __forceinline__ float fast_sigmoid(float x) {
    return 0.5f * fast_tanh(0.5f * x) + 0.5f;
}

#if HAVE_TDM
// Issue TENSOR_LOAD_TO_LDS of a 2-D fp32 tile: tile_d0 contiguous elems per row,
// tile_d1 rows, row stride 'stride0' elems, into LDS at ldsOff.
__device__ __forceinline__ void tdm_load_2d_f32(const void* gptr, unsigned ldsOff,
                                                int tile_d0, int tile_d1, int stride0) {
    const unsigned long long ga = (unsigned long long)gptr;
    u32x4 g0;
    g0[0] = 1u;                                        // count=1, user descriptor
    g0[1] = ldsOff;                                    // lds_addr (bytes)
    g0[2] = (unsigned)ga;                              // global_addr[31:0]
    g0[3] = (unsigned)((ga >> 32) & 0x01FFFFFFu) | (2u << 30);  // addr[56:32] | type=2
    // Group1: data_size=4B (code 2); generous tensor dims; tile dims; dim0 stride.
    const unsigned td0 = 1u << 24, td1 = 1u << 24;     // large => no OOB clipping
    i32x8 g1;
    g1[0] = 0x20000;                                   // workgroup_mask=0, data_size=2
    g1[1] = (int)((td0 & 0xFFFFu) << 16);              // tensor_dim0[15:0] @ bits 47:32? (79:48 low)
    g1[2] = (int)((td0 >> 16) | ((td1 & 0xFFFFu) << 16));
    g1[3] = (int)((td1 >> 16) | ((unsigned)tile_d0 << 16));   // tile_dim0 @127:112
    g1[4] = tile_d1;                                   // tile_dim1 @143:128 (tile_dim2=0)
    g1[5] = stride0;                                   // tensor_dim0_stride[31:0]
    g1[6] = 0;                                         // stride hi / dim1_stride lo
    g1[7] = 0;
    i32x4 gz = {0, 0, 0, 0};
#if __clang_major__ >= 23
    i32x8 gz8 = {0, 0, 0, 0, 0, 0, 0, 0};
    __builtin_amdgcn_tensor_load_to_lds(g0, g1, gz, gz, gz8, 0);
#else
    __builtin_amdgcn_tensor_load_to_lds(g0, g1, gz, gz, 0);
#endif
}
#endif

// ---- LDS overlay layout inside regionB ----
#define XBYTES      (RPB * T_LEN * 4)                 // 32768: x slice [row][t] f32
#define HOFF        XBYTES                            // h ping-pong (f16) 2*16*64*2 = 4096
#define POFF        (HOFF + 2 * RPB * HDIM * 2)       // fc partials
#define RB_RUNTIME  (POFF + RPB * 8 * 4)
#define RB_INIT     (GDIM * HDIM * 4)                 // 65536: fp32 W_hh staging (TDM path)
#if HAVE_TDM
#define RB_SIZE     (RB_INIT > RB_RUNTIME ? RB_INIT : RB_RUNTIME)
#else
#define RB_SIZE     RB_RUNTIME
#endif

__launch_bounds__(NTHR)
__global__ void lstm_wmma_kernel(const float* __restrict__ x,
                                 const float* __restrict__ W_ih,
                                 const float* __restrict__ W_hh,
                                 const float* __restrict__ b_ih,
                                 const float* __restrict__ b_hh,
                                 const float* __restrict__ W_fc,
                                 const float* __restrict__ b_fc,
                                 float* __restrict__ out) {
    __shared__ _Float16 Wlds[GDIM * HDIM];                 // persistent f16 W_hh [n][k], 32KB
    __shared__ __align__(16) unsigned char regionB[RB_SIZE];

    float*    xlds  = (float*)regionB;                     // [row][t], 32KB
    _Float16* hbufA = (_Float16*)(regionB + HOFF);         // ping
    _Float16* hbufB = hbufA + RPB * HDIM;                  // pong
    float*    part  = (float*)(regionB + POFF);

    const int tid  = threadIdx.x;
    const int lane = tid & 31;
    const int wave = tid >> 5;            // 0..3 = h-column slice
    const int cs   = wave;
    const int cl   = lane & 15;
    const int hi   = (lane >> 4) & 1;
    const int row0 = blockIdx.x * RPB;

#if HAVE_TDM
    // ---- Stage 1: TDM-load W_hh (256x64 f32 contiguous) into regionB ----
    if (wave == 0) {
        tdm_load_2d_f32(W_hh, (unsigned)(size_t)(void*)regionB, HDIM, GDIM, HDIM);
        __builtin_amdgcn_s_wait_tensorcnt(0);
    }
    __syncthreads();
    {   // convert fp32 staging -> persistent f16 Wlds
        const float* Wf32 = (const float*)regionB;
        for (int i = tid; i < GDIM * HDIM; i += NTHR)
            Wlds[i] = (_Float16)Wf32[i];
    }
    __syncthreads();
    // ---- Stage 2: TDM-load this block's x slice [16 rows][512 t] f32 ----
    if (wave == 0) {
        tdm_load_2d_f32(&x[(size_t)row0 * T_LEN], (unsigned)(size_t)(void*)regionB,
                        T_LEN, RPB, T_LEN);
        __builtin_amdgcn_s_wait_tensorcnt(0);
    }
    for (int i = tid; i < RPB * HDIM; i += NTHR)           // zero h ping (doesn't overlap x)
        hbufA[i] = (_Float16)0.0f;
#else
    for (int i = tid; i < GDIM * HDIM; i += NTHR)
        Wlds[i] = (_Float16)W_hh[i];
    {   // coalesced float4 staging of x slice, layout [row][t]
        const int nf4 = RPB * T_LEN / 4;                   // 2048
        for (int i = tid; i < nf4; i += NTHR) {
            int r  = i >> 7;                               // 128 float4 per row
            int c4 = i & 127;
            float4 v = *(const float4*)&x[(size_t)(row0 + r) * T_LEN + c4 * 4];
            *(float4*)&xlds[r * T_LEN + c4 * 4] = v;
        }
    }
    for (int i = tid; i < RPB * HDIM; i += NTHR)
        hbufA[i] = (_Float16)0.0f;
    __syncthreads();                                       // Wlds ready before B-frag reads
#endif

    // ---- per-lane constants: this lane owns gate columns q*64 + nh ----
    const int nh = cs * 16 + cl;          // h column 0..63
    float wih[4], bias[4];
#pragma unroll
    for (int q = 0; q < 4; ++q) {
        int g = q * 64 + nh;              // torch gate order i,f,g,o
        wih[q]  = W_ih[g];                // I == 1
        bias[q] = b_ih[g] + b_hh[g];
    }

    // ---- B fragments (W_hh^T) resident in registers ----
    V16 bfrag[4][2];
#pragma unroll
    for (int q = 0; q < 4; ++q) {
        const int nabs = q * 64 + nh;
#pragma unroll
        for (int kh = 0; kh < 2; ++kh) {
            const _Float16* p = &Wlds[nabs * HDIM + kh * 32 + hi * 16];
            bfrag[q][kh].p.lo = *(const v8h*)(p);
            bfrag[q][kh].p.hi = *(const v8h*)(p + 8);
        }
    }
    __syncthreads();                                       // x / h buffers ready

    float creg[8], hreg[8];
#pragma unroll
    for (int r = 0; r < 8; ++r) { creg[r] = 0.0f; hreg[r] = 0.0f; }

    _Float16* hin  = hbufA;
    _Float16* hout = hbufB;

    for (int t = 0; t < T_LEN; ++t) {
        // A fragments: h[cl][*] f16 (A layout: lane<16 K={0..15}, lane>=16 K={16..31})
        V16 afrag[2];
        const _Float16* hrow = &hin[cl * HDIM];
#pragma unroll
        for (int kh = 0; kh < 2; ++kh) {
            const int base = kh * 32 + hi * 16;
            afrag[kh].p.lo = *(const v8h*)(hrow + base);
            afrag[kh].p.hi = *(const v8h*)(hrow + base + 8);
        }

        v8f acc[4];
#pragma unroll
        for (int q = 0; q < 4; ++q) {
            v8f c0 = {};
            c0 = __builtin_amdgcn_wmma_f32_16x16x32_f16(false, afrag[0].v, false, bfrag[q][0].v,
                                                        (short)0, c0, false, false);
            c0 = __builtin_amdgcn_wmma_f32_16x16x32_f16(false, afrag[1].v, false, bfrag[q][1].v,
                                                        (short)0, c0, false, false);
            acc[q] = c0;
        }

        float xv[8];
#pragma unroll
        for (int r = 0; r < 8; ++r)
            xv[r] = xlds[(hi * 8 + r) * T_LEN + t];        // broadcast LDS reads

#pragma unroll
        for (int r = 0; r < 8; ++r) {
            const float iv = fast_sigmoid(acc[0][r] + xv[r] * wih[0] + bias[0]);
            const float fv = fast_sigmoid(acc[1][r] + xv[r] * wih[1] + bias[1]);
            const float gv = fast_tanh   (acc[2][r] + xv[r] * wih[2] + bias[2]);
            const float ov = fast_sigmoid(acc[3][r] + xv[r] * wih[3] + bias[3]);
            const float cn = fv * creg[r] + iv * gv;
            creg[r] = cn;
            const float hn = ov * fast_tanh(cn);
            hreg[r] = hn;
            hout[(hi * 8 + r) * HDIM + nh] = (_Float16)hn;
        }
        __syncthreads();
        { _Float16* tmp = hin; hin = hout; hout = tmp; }
    }

    // ---- fc(h_T): deterministic fp32 cross-lane reduction ----
    const float wfc0 = W_fc[0 * HDIM + nh];
    const float wfc1 = W_fc[1 * HDIM + nh];
#pragma unroll
    for (int r = 0; r < 8; ++r) {
        float s0 = hreg[r] * wfc0;
        float s1 = hreg[r] * wfc1;
#pragma unroll
        for (int m = 8; m >= 1; m >>= 1) {
            s0 += __shfl_xor(s0, m, 32);
            s1 += __shfl_xor(s1, m, 32);
        }
        if (cl == 0) {                     // lanes 0 / 16 hold the two row sums
            const int rowl = hi * 8 + r;
            part[rowl * 8 + cs * 2 + 0] = s0;
            part[rowl * 8 + cs * 2 + 1] = s1;
        }
    }
    __syncthreads();
    if (tid < RPB * 2) {
        const int rowl = tid >> 1;
        const int cc   = tid & 1;
        float s = b_fc[cc];
#pragma unroll
        for (int c2 = 0; c2 < 4; ++c2) s += part[rowl * 8 + c2 * 2 + cc];
        out[(size_t)(row0 + rowl) * 2 + cc] = s;
    }
}

extern "C" void kernel_launch(void* const* d_in, const int* in_sizes, int n_in,
                              void* d_out, int out_size, void* d_ws, size_t ws_size,
                              hipStream_t stream) {
    (void)in_sizes; (void)n_in; (void)out_size; (void)d_ws; (void)ws_size;
    const float* x    = (const float*)d_in[0];
    const float* W_ih = (const float*)d_in[1];
    const float* W_hh = (const float*)d_in[2];
    const float* b_ih = (const float*)d_in[3];
    const float* b_hh = (const float*)d_in[4];
    const float* W_fc = (const float*)d_in[5];
    const float* b_fc = (const float*)d_in[6];
    float* out = (float*)d_out;

    dim3 grid(B_TOT / RPB);   // 128 blocks -> many WGPs busy
    dim3 block(NTHR);         // 4 waves (wave32)
    hipLaunchKernelGGL(lstm_wmma_kernel, grid, block, 0, stream,
                       x, W_ih, W_hh, b_ih, b_hh, W_fc, b_fc, out);
}
